// MultiHeadSelfAttention_49520972923383
// MI455X (gfx1250) — matrix operand order repaired and checked by reading, not compiled, not run on hardware
//
#include <hip/hip_runtime.h>
#include <hip/hip_bf16.h>

typedef __bf16 bf16_t;
typedef __attribute__((ext_vector_type(16))) __bf16 v16bf;
typedef __attribute__((ext_vector_type(8)))  __bf16 v8bf;
typedef __attribute__((ext_vector_type(4)))  __bf16 v4bf;
typedef __attribute__((ext_vector_type(8)))  float  v8f;

#define EMBED 1024
#define HEADS 16
#define HDIM  64
#define BATCH 8
#define SEQ   1024
#define MTOT  (BATCH * SEQ)   // 8192 tokens

// ---------------------------------------------------------------------------
// A-fragment builder: 16x32 bf16 tile, contiguous-k A layout (same k order as the B loads).
// lane holds row m; elems 0..15 -> K=+16*lg+j (contiguous)
// ---------------------------------------------------------------------------
__device__ __forceinline__ v16bf load_afrag(const bf16_t* row, int kb, int lg) {
  v8bf a0 = *(const v8bf*)(row + kb + 16 * lg);
  v8bf a1 = *(const v8bf*)(row + kb + 16 * lg + 8);
  v16bf a;
#pragma unroll
  for (int j = 0; j < 8; ++j) { a[j] = a0[j]; a[j + 8] = a1[j]; }
  return a;
}

// CDNA5 async copy engine: 16B per lane, global -> LDS, tracked by ASYNCcnt.
// GVS addressing: saddr(SGPR pair) + 32-bit per-lane byte offset.
__device__ __forceinline__ void async_cp_b128(uint32_t lds_addr,
                                              uint32_t byte_off,
                                              const void* gbase) {
  asm volatile("global_load_async_to_lds_b128 %0, %1, %2"
               :: "v"(lds_addr), "v"(byte_off), "s"(gbase)
               : "memory");
}

// ---------------------------------------------------------------------------
// fp32 -> bf16 conversion (4 elements / thread)
// ---------------------------------------------------------------------------
__global__ void cvt_f32_bf16(const float* __restrict__ in,
                             bf16_t* __restrict__ out, int n4) {
  int i = blockIdx.x * blockDim.x + threadIdx.x;
  if (i < n4) {
    float4 f = ((const float4*)in)[i];
    v4bf o;
    o[0] = (bf16_t)f.x; o[1] = (bf16_t)f.y;
    o[2] = (bf16_t)f.z; o[3] = (bf16_t)f.w;
    ((v4bf*)out)[i] = o;
  }
}

// ---------------------------------------------------------------------------
// QKV projection:  out = x @ W^T + bias   (bf16 in, f32 accum, bf16 out)
// Block = 8 waves, tile 128x64. Double-buffered LDS staging filled with
// global_load_async_to_lds_b128; s_wait_asynccnt 3 keeps the next k-step's
// 3 copies in flight. Buffers are explicit swapped variables and the loop is
// kept un-unrolled so accumulators stay pinned (no copy/NOP churn); all DS
// fragment reads are batched so there is ONE s_wait_dscnt per k-step.
// vmode==0: store per-(b,h) as [L,64] (Q,K). vmode==1: transposed [64,L] (V).
// ---------------------------------------------------------------------------
__global__ void __launch_bounds__(256)
qkv_gemm(const bf16_t* __restrict__ X,   // [8192,1024]
         const bf16_t* __restrict__ W,   // [1024,1024]
         const float*  __restrict__ bias,
         bf16_t* __restrict__ out, int vmode) {
  __shared__ __align__(16) bf16_t xs[2][128 * 32];   // 2 x 8KB
  __shared__ __align__(16) bf16_t wls[2][64 * 32];   // 2 x 4KB

  const int tid  = threadIdx.x;
  const int lane = tid & 31;
  const int wave = tid >> 5;
  const int lg = lane >> 4, ln = lane & 15;
  const int mbase = (blockIdx.x >> 4) * 128;   // 64 m-tiles
  const int nbase = (blockIdx.x & 15) * 64;    // 16 n-groups

  // async-copy geometry: 16B chunks; X tile = 512 chunks (2/thread),
  // W tile = 256 chunks (1/thread).
  const int crow = tid >> 2;                   // 0..63
  const int ckc  = (tid & 3) * 8;              // 0,8,16,24
  const uint32_t xgo0 = (uint32_t)(((mbase + crow) * EMBED + ckc) * 2);
  const uint32_t xgo1 = (uint32_t)(((mbase + 64 + crow) * EMBED + ckc) * 2);
  const uint32_t wgo  = (uint32_t)(((nbase + crow) * EMBED + ckc) * 2);

  // copy-destination LDS addrs + read-base pointers, double-buffered via swap
  uint32_t xc0_c = (uint32_t)(uintptr_t)&xs[0][crow * 32 + ckc];
  uint32_t xc1_c = (uint32_t)(uintptr_t)&xs[0][(64 + crow) * 32 + ckc];
  uint32_t wc_c  = (uint32_t)(uintptr_t)&wls[0][crow * 32 + ckc];
  uint32_t xc0_n = (uint32_t)(uintptr_t)&xs[1][crow * 32 + ckc];
  uint32_t xc1_n = (uint32_t)(uintptr_t)&xs[1][(64 + crow) * 32 + ckc];
  uint32_t wc_n  = (uint32_t)(uintptr_t)&wls[1][crow * 32 + ckc];
  const bf16_t* xr_c = &xs[0][(wave * 16 + ln) * 32];
  const bf16_t* xr_n = &xs[1][(wave * 16 + ln) * 32];
  const bf16_t* wr_c = &wls[0][ln * 32 + 16 * lg];
  const bf16_t* wr_n = &wls[1][ln * 32 + 16 * lg];

  v8f acc[4] = {};

  // prologue: fill buffer 0 with k-step 0
  async_cp_b128(xc0_c, xgo0, X);
  async_cp_b128(xc1_c, xgo1, X);
  async_cp_b128(wc_c,  wgo,  W);

#pragma clang loop unroll(disable)
  for (int kb = 0; kb < EMBED; kb += 32) {
    const uint32_t ko = (uint32_t)(((kb + 32 < EMBED) ? kb + 32 : 0) * 2);
    // issue next k-step into the other buffer (last iter: harmless redo of 0)
    async_cp_b128(xc0_n, xgo0 + ko, X);
    async_cp_b128(xc1_n, xgo1 + ko, X);
    async_cp_b128(wc_n,  wgo + ko,  W);
    // current buffer's 3 copies done; next buffer's 3 stay in flight
    asm volatile("s_wait_asynccnt 0x3" ::: "memory");
    __syncthreads();

    // batch ALL fragment reads (2 + 8 ds_load_b128), then 4 back-to-back WMMAs
    v16bf a = load_afrag(xr_c, 0, lg);
    v16bf b[4];
#pragma unroll
    for (int t = 0; t < 4; ++t) b[t] = *(const v16bf*)(wr_c + t * 512);
#pragma unroll
    for (int t = 0; t < 4; ++t)
      acc[t] = __builtin_amdgcn_wmma_f32_16x16x32_bf16(
          false, a, false, b[t], (short)0, acc[t], false, false);
    __syncthreads();   // all waves done reading cur before it is refilled

    // swap double buffers (scalar-cheap; loop body stays invariant)
    uint32_t ti;
    ti = xc0_c; xc0_c = xc0_n; xc0_n = ti;
    ti = xc1_c; xc1_c = xc1_n; xc1_n = ti;
    ti = wc_c;  wc_c  = wc_n;  wc_n  = ti;
    const bf16_t* tp;
    tp = xr_c; xr_c = xr_n; xr_n = tp;
    tp = wr_c; wr_c = wr_n; wr_n = tp;
  }

#pragma unroll
  for (int t = 0; t < 4; ++t) {
    const int col = nbase + t * 16 + ln;
    const float bv = bias[col];
    const int h = col >> 6, dd = col & 63;
#pragma unroll
    for (int r = 0; r < 8; ++r) {
      const int row = mbase + wave * 16 + r + 8 * lg;  // C layout: VGPR r->M
      const int b = row >> 10, l = row & 1023;
      const float v = acc[t][r] + bv;
      size_t idx;
      if (vmode)  // V transposed: [(b*16+h)][dd][l]
        idx = ((size_t)(b * HEADS + h) * HDIM + dd) * SEQ + l;
      else        // Q/K: [(b*16+h)][l][dd]
        idx = ((size_t)(b * HEADS + h) * SEQ + l) * HDIM + dd;
      out[idx] = (bf16_t)v;
    }
  }
}

// ---------------------------------------------------------------------------
// Flash attention: one wave = 16 queries of one (b,h); 4 waves per block.
// Per 32-key step: 4 WMMAs (scores) + online softmax + 4 WMMAs (PV).
// K fragments prefetched one step ahead; V loads issued before the softmax
// VALU block so their latency hides under the exp math.
// ---------------------------------------------------------------------------
__global__ void attn(const bf16_t* __restrict__ Q,   // [128][1024][64]
                     const bf16_t* __restrict__ K,   // [128][1024][64]
                     const bf16_t* __restrict__ Vt,  // [128][64][1024]
                     float* __restrict__ O) {        // [128][1024][64]
  __shared__ __align__(16) bf16_t plds[4][16][32];   // per-wave P staging

  const int lane = threadIdx.x & 31;
  const int wave = threadIdx.x >> 5;
  const int bh = blockIdx.x >> 4;
  const int qbase = ((blockIdx.x & 15) << 6) + wave * 16;
  const int lg = lane >> 4, ln = lane & 15;

  const bf16_t* Qb = Q + (size_t)bh * SEQ * HDIM;
  const bf16_t* Kb = K + (size_t)bh * SEQ * HDIM;
  const bf16_t* Vb = Vt + (size_t)bh * HDIM * SEQ;

  v16bf aq[2];
  {
    const bf16_t* qrow = Qb + (size_t)(qbase + ln) * HDIM;
#pragma unroll
    for (int s = 0; s < 2; ++s) aq[s] = load_afrag(qrow, s * 32, lg);
  }

  const bf16_t* kbase0 = Kb + (size_t)ln * HDIM + 16 * lg;
  const bf16_t* vbase[4];
#pragma unroll
  for (int t = 0; t < 4; ++t)
    vbase[t] = Vb + (size_t)(t * 16 + ln) * SEQ + 16 * lg;

  float mrow[8], lrow[8];
#pragma unroll
  for (int r = 0; r < 8; ++r) { mrow[r] = -__builtin_inff(); lrow[r] = 0.f; }
  v8f oacc[4] = {};

  v16bf bk_n[4];
#pragma unroll
  for (int h2 = 0; h2 < 2; ++h2)
#pragma unroll
    for (int sd = 0; sd < 2; ++sd)
      bk_n[h2 * 2 + sd] =
          *(const v16bf*)(kbase0 + (size_t)(h2 * 16) * HDIM + sd * 32);

  for (int kb = 0; kb < SEQ; kb += 32) {
    v16bf bk_c[4];
#pragma unroll
    for (int f = 0; f < 4; ++f) bk_c[f] = bk_n[f];

    v16bf bv_c[4];
#pragma unroll
    for (int t = 0; t < 4; ++t) bv_c[t] = *(const v16bf*)(vbase[t] + kb);

    const int kn = (kb + 32 < SEQ) ? kb + 32 : 0;
#pragma unroll
    for (int h2 = 0; h2 < 2; ++h2)
#pragma unroll
      for (int sd = 0; sd < 2; ++sd)
        bk_n[h2 * 2 + sd] =
            *(const v16bf*)(kbase0 + (size_t)(kn + h2 * 16) * HDIM + sd * 32);

    v8f s2[2] = {};
#pragma unroll
    for (int h2 = 0; h2 < 2; ++h2)
#pragma unroll
      for (int sd = 0; sd < 2; ++sd)
        s2[h2] = __builtin_amdgcn_wmma_f32_16x16x32_bf16(
            false, aq[sd], false, bk_c[h2 * 2 + sd], (short)0, s2[h2],
            false, false);

    float pv[2][8];
#pragma unroll
    for (int r = 0; r < 8; ++r) {
      float a0 = s2[0][r] * 0.125f;   // d^-0.5 = 1/8
      float a1 = s2[1][r] * 0.125f;
      float mx = fmaxf(a0, a1);
#pragma unroll
      for (int msk = 8; msk >= 1; msk >>= 1)
        mx = fmaxf(mx, __shfl_xor(mx, msk, 32));
      const float mnew = fmaxf(mrow[r], mx);
      const float corr = __expf(mrow[r] - mnew);
      const float p0 = __expf(a0 - mnew);
      const float p1 = __expf(a1 - mnew);
      float rs = p0 + p1;
#pragma unroll
      for (int msk = 8; msk >= 1; msk >>= 1)
        rs += __shfl_xor(rs, msk, 32);
      lrow[r] = lrow[r] * corr + rs;
      mrow[r] = mnew;
      pv[0][r] = p0; pv[1][r] = p1;
#pragma unroll
      for (int t = 0; t < 4; ++t) oacc[t][r] *= corr;
    }

    __syncthreads();
#pragma unroll
    for (int h2 = 0; h2 < 2; ++h2)
#pragma unroll
      for (int r = 0; r < 8; ++r)
        plds[wave][r + 8 * lg][h2 * 16 + ln] = (bf16_t)pv[h2][r];
    __syncthreads();

    v16bf ap;
    {
      v8bf p0 = *(const v8bf*)&plds[wave][ln][16 * lg];
      v8bf p1 = *(const v8bf*)&plds[wave][ln][16 * lg + 8];
#pragma unroll
      for (int j = 0; j < 8; ++j) { ap[j] = p0[j]; ap[j + 8] = p1[j]; }
    }

#pragma unroll
    for (int t = 0; t < 4; ++t)
      oacc[t] = __builtin_amdgcn_wmma_f32_16x16x32_bf16(
          false, ap, false, bv_c[t], (short)0, oacc[t], false, false);
  }

#pragma unroll
  for (int r = 0; r < 8; ++r) {
    const float inv = 1.0f / lrow[r];
    const int q = qbase + r + 8 * lg;
#pragma unroll
    for (int t = 0; t < 4; ++t)
      O[((size_t)bh * SEQ + q) * HDIM + t * 16 + ln] = oacc[t][r] * inv;
  }
}

// ---------------------------------------------------------------------------
extern "C" void kernel_launch(void* const* d_in, const int* in_sizes, int n_in,
                              void* d_out, int out_size, void* d_ws, size_t ws_size,
                              hipStream_t stream) {
  const float* x  = (const float*)d_in[0];
  const float* wq = (const float*)d_in[1];
  const float* bq = (const float*)d_in[2];
  const float* wk = (const float*)d_in[3];
  const float* bk = (const float*)d_in[4];
  const float* wv = (const float*)d_in[5];
  const float* bv = (const float*)d_in[6];
  float* out = (float*)d_out;

  // workspace carve-up (bf16): x 16MB, W 3x2MB, Q/K/Vt 3x16MB = 70MB
  bf16_t* Xb  = (bf16_t*)d_ws;
  bf16_t* Wqb = Xb  + (size_t)MTOT * EMBED;
  bf16_t* Wkb = Wqb + (size_t)EMBED * EMBED;
  bf16_t* Wvb = Wkb + (size_t)EMBED * EMBED;
  bf16_t* Qb  = Wvb + (size_t)EMBED * EMBED;
  bf16_t* Kb  = Qb  + (size_t)MTOT * EMBED;
  bf16_t* Vtb = Kb  + (size_t)MTOT * EMBED;

  const int n4x = (MTOT * EMBED) / 4;
  const int n4w = (EMBED * EMBED) / 4;
  cvt_f32_bf16<<<(n4x + 255) / 256, 256, 0, stream>>>(x,  Xb,  n4x);
  cvt_f32_bf16<<<(n4w + 255) / 256, 256, 0, stream>>>(wq, Wqb, n4w);
  cvt_f32_bf16<<<(n4w + 255) / 256, 256, 0, stream>>>(wk, Wkb, n4w);
  cvt_f32_bf16<<<(n4w + 255) / 256, 256, 0, stream>>>(wv, Wvb, n4w);

  // 64 m-tiles x 16 n-groups = 1024 blocks, 8 waves each
  qkv_gemm<<<1024, 256, 0, stream>>>(Xb, Wqb, bq, Qb,  0);
  qkv_gemm<<<1024, 256, 0, stream>>>(Xb, Wkb, bk, Kb,  0);
  qkv_gemm<<<1024, 256, 0, stream>>>(Xb, Wvb, bv, Vtb, 1);

  // 128 (b,h) x 16 q-groups = 2048 blocks, 4 waves (64 queries) each
  attn<<<2048, 128, 0, stream>>>(Qb, Kb, Vtb, out);
}
